// ContrastDrivenFeatureAggregation_37615323578798
// MI455X (gfx1250) — matrix operand order repaired and checked by reading, not compiled, not run on hardware
//
#include <hip/hip_runtime.h>

// MI455X / gfx1250, wave32, WMMA path.
// Pipeline: conv1 -> conv2 -> v-linear -> attn-linear -> softmax+agg+fold -> proj -> conv3 -> conv4
// All conv/linear layers share one implicit-GEMM WMMA kernel (f16 A/B, f32 accumulate).
// Conv weights pre-permuted to r-major K order (K' = r*Cin + c): each 32-wide K-step
// has a uniform 3x3 tap -> scalar-only step state, no per-lane division.
// LDS double-buffered: one barrier per K-step, next step's global loads overlap WMMAs.

typedef __attribute__((ext_vector_type(16))) _Float16 v16h;
typedef __attribute__((ext_vector_type(8)))  _Float16 v8h;
typedef __attribute__((ext_vector_type(8)))  float    v8f;

#define HH 96
#define WW 96
#define LL (HH * WW)          // 9216 pixels
#define BATCH 8
#define NB 64                 // pixel tile (N) -> 4 WMMAs per wave per K-step
#define MBK 128               // M tile
#define KBS 32                // K step
#define TST 40                // LDS row stride in halfs (keeps fragment reads 16B-aligned)

// ---------------------------------------------------------------------------
// Weight prep kernels (run once per launch; weights are tiny).
// ---------------------------------------------------------------------------
__global__ __launch_bounds__(256)
void cvt_conv_w_kernel(const float* __restrict__ s, _Float16* __restrict__ d,
                       int M, int Cin) {
    int i = blockIdx.x * 256 + threadIdx.x;
    int tot = M * Cin * 9;
    if (i >= tot) return;
    int m   = i / (Cin * 9);
    int rem = i % (Cin * 9);
    int c   = rem / 9;
    int r   = rem % 9;
    d[((size_t)m * 9 + r) * Cin + c] = (_Float16)s[i];
}

__global__ __launch_bounds__(256)
void cvt_pad_kernel(const float* __restrict__ s, _Float16* __restrict__ d,
                    int nsrc, int ntot) {
    int i = blockIdx.x * 256 + threadIdx.x;
    if (i < ntot) d[i] = (i < nsrc) ? (_Float16)s[i] : (_Float16)0.f;
}

// ---------------------------------------------------------------------------
// Implicit-GEMM conv / linear with WMMA.
//   out[b][m][l] = act( scale[m] * (sum_k W[m][k] * im2col(in)[k][l]) + bias[m] )
//   KHW = 9 : 3x3 conv, pad 1 (weights r-major).   KHW = 1 : pointwise.
// Block: 256 threads (8 waves). Tile: 128 M x 64 pixels. Grid: (L/64, ceil(M/128), B).
// Weight matrix must have >= gridDim.y*128 rows (zero padded).
// ---------------------------------------------------------------------------
template<int KHW, bool BN, bool RELU>
__global__ __launch_bounds__(256)
void gemm_conv_kernel(const float* __restrict__ in,
                      const _Float16* __restrict__ wgt,   // [Mpad][Ktot] f16
                      const float* __restrict__ gam,
                      const float* __restrict__ bet,
                      float* __restrict__ out,            // [B][M][L] f32
                      int Cin, int M)
{
    __shared__ _Float16 At[2][MBK][TST];   // double-buffered weight tile
    __shared__ _Float16 Bt[2][NB][TST];    // double-buffered im2col tile

    const int tid  = threadIdx.x;
    const int lane = tid & 31;
    const int wave = tid >> 5;
    const int b    = blockIdx.z;
    const int nb0  = blockIdx.x * NB;
    const int mb0  = blockIdx.y * MBK;
    const int Ktot = Cin * KHW;
    const int nk   = Ktot / KBS;

    // B staging role: fixed pixel, 8 contiguous K slots
    const int pp  = tid & 63;
    const int kl8 = (tid >> 6) * 8;        // 0,8,16,24
    const int n_g = nb0 + pp;
    const int py  = n_g / WW;
    const int px  = n_g % WW;
    const float* thr_base = in + (size_t)b * Cin * LL + (size_t)kl8 * LL + n_g;

    // per-thread 3x3 tap validity mask (bit r set if tap r is in-bounds)
    int vmask = 0x1FF;
    if (KHW == 9) {
        vmask = 0;
        #pragma unroll
        for (int r = 0; r < 9; ++r) {
            const int yy = py + r / 3 - 1, xx = px + r % 3 - 1;
            if (yy >= 0 && yy < HH && xx >= 0 && xx < WW) vmask |= 1 << r;
        }
    }

    // A staging role: row tid/2, 16-half column chunk (unconditional: rows padded)
    const int wm = tid >> 1;
    const int wc = (tid & 1) * 16;
    const _Float16* wrow = wgt + (size_t)(mb0 + wm) * Ktot + wc;

    // scalar state for the NEXT K-step to be loaded
    int cbase = 0, rIdx = 0;
    int dy0 = (KHW == 9) ? -1 : 0, dx0 = (KHW == 9) ? -1 : 0;

    auto load_step = [&](int kbase, v8h& w0, v8h& w1, v8h& bv) {
        w0 = *(const v8h*)(wrow + kbase);
        w1 = *(const v8h*)(wrow + kbase + 8);
        __builtin_prefetch(wrow + kbase + KBS, 0, 1);

        const bool ok = (KHW == 1) || ((vmask >> rIdx) & 1);
        const int soff = cbase * LL + dy0 * WW + dx0;       // uniform scalar offset
        const float* src = thr_base + soff;
        float fv[8];
        #pragma unroll
        for (int i = 0; i < 8; ++i) fv[i] = 0.f;
        if (ok) {                                            // one exec-masked clause
            #pragma unroll
            for (int i = 0; i < 8; ++i) fv[i] = src[(size_t)i * LL];
        }
        #pragma unroll
        for (int i = 0; i < 8; ++i) bv[i] = (_Float16)fv[i];

        // advance scalar step state
        cbase += KBS;
        if (KHW == 9 && cbase == Cin) {
            cbase = 0;
            ++rIdx;
            if (++dx0 == 2) { dx0 = -1; ++dy0; }
        }
    };

    v8f acc[4] = {{}, {}, {}, {}};
    v8h w0, w1, bv;
    load_step(0, w0, w1, bv);                // prologue: step 0 into registers

    for (int kb = 0; kb < nk; ++kb) {
        const int p = kb & 1;

        // ---- stage current step into LDS buffer p ----
        *(v8h*)&At[p][wm][wc]     = w0;
        *(v8h*)&At[p][wm][wc + 8] = w1;
        *(v8h*)&Bt[p][pp][kl8]    = bv;      // one ds_store_b128
        __syncthreads();

        // ---- issue next step's global loads (overlap with WMMAs below) ----
        if (kb + 1 < nk) load_step((kb + 1) * KBS, w0, w1, bv);

        // ---- A fragment (ISA 16-bit A 16x32 layout) ----
        const int am = wave * 16 + (lane & 15);
        const int ak = (lane >> 4) * 8;      // lanes 0-15: K0..7/16..23; 16-31: K8..15/24..31
        v8h alo = *(const v8h*)&At[p][am][ak];
        v8h ahi = *(const v8h*)&At[p][am][ak + 16];
        v16h a;
        #pragma unroll
        for (int i = 0; i < 8; ++i) { a[i] = alo[i]; a[i + 8] = ahi[i]; }

        // ---- B fragments: 4 N-subtiles of 16 pixels ----
        const int bk = (lane >> 4) * 8;
        const int bn = lane & 15;
        #pragma unroll
        for (int j = 0; j < 4; ++j) {
            v8h blo = *(const v8h*)&Bt[p][j * 16 + bn][bk];
            v8h bhi = *(const v8h*)&Bt[p][j * 16 + bn][bk + 16];
            v16h f;
            #pragma unroll
            for (int i = 0; i < 8; ++i) { f[i] = blo[i]; f[i + 8] = bhi[i]; }
            acc[j] = __builtin_amdgcn_wmma_f32_16x16x32_f16(false, a, false, f,
                                                            (short)0, acc[j], false, false);
        }
    }

    // ---- epilogue: C/D layout -> N = lane&15, M = r + 8*(lane>>4) ----
    #pragma unroll
    for (int r = 0; r < 8; ++r) {
        const int mg = mb0 + wave * 16 + (lane >> 4) * 8 + r;
        if (mg < M) {
            float s = 1.f, o = 0.f;
            if (BN) { s = gam[mg] * rsqrtf(1.0f + 1e-5f); o = bet[mg]; }
            else if (bet) { o = bet[mg]; }
            float* orow = out + ((size_t)b * M + mg) * LL + nb0 + (lane & 15);
            #pragma unroll
            for (int j = 0; j < 4; ++j) {
                float y = acc[j][r] * s + o;
                if (RELU) y = fmaxf(y, 0.f);
                orow[j * 16] = y;
            }
        }
    }
}

// ---------------------------------------------------------------------------
// Fused softmax + local attention einsum + fold.
// folded[b, h*32+d, y, x] = sum_{p=(ki,kj)} sum_q
//     softmax_q(attn[b, h*81+p*9+q, l'] * SCALE) * v[b, h*32+d, y'+qi-1, x'+qj-1]
// with (y', x') = (y+1-ki, x+1-kj).  Each (l', p, q, d) term is used exactly once,
// so the fold is a pure gather: no xw tensor, no redundant softmax.
// ---------------------------------------------------------------------------
__global__ __launch_bounds__(256)
void agg_fold_kernel(const float* __restrict__ attn,   // [B][324][L]
                     const float* __restrict__ v,      // [B][128][L]
                     float* __restrict__ out)          // [B][128][L]
{
    const int t = blockIdx.x * blockDim.x + threadIdx.x;
    const int total = BATCH * 4 * LL;
    if (t >= total) return;
    const int l = t % LL;
    const int h = (t / LL) & 3;
    const int b = t / (4 * LL);
    const int y = l / WW, x = l % WW;
    const float SCALE = 0.17677669529663687f;   // 32^-0.5

    float acc[32];
    #pragma unroll
    for (int d = 0; d < 32; ++d) acc[d] = 0.f;

    for (int p = 0; p < 9; ++p) {
        const int ki = p / 3, kj = p % 3;
        const int yp = y + 1 - ki, xp = x + 1 - kj;
        if (yp < 0 || yp >= HH || xp < 0 || xp >= WW) continue;
        const int lp = yp * WW + xp;

        const float* ab = attn + ((size_t)b * 324 + h * 81 + p * 9) * LL + lp;
        float aq[9], mx = -1e30f;
        #pragma unroll
        for (int q = 0; q < 9; ++q) { aq[q] = ab[(size_t)q * LL] * SCALE; mx = fmaxf(mx, aq[q]); }
        float sum = 0.f;
        #pragma unroll
        for (int q = 0; q < 9; ++q) { aq[q] = __expf(aq[q] - mx); sum += aq[q]; }
        const float inv = 1.f / sum;

        #pragma unroll
        for (int q = 0; q < 9; ++q) {
            const int yv = yp + q / 3 - 1, xv = xp + q % 3 - 1;
            if (yv < 0 || yv >= HH || xv < 0 || xv >= WW) continue;
            const float wq = aq[q] * inv;
            const float* vb = v + (((size_t)b * 128 + h * 32) * HH + yv) * WW + xv;
            #pragma unroll
            for (int d = 0; d < 32; ++d)
                acc[d] += wq * vb[(size_t)d * LL];
        }
    }

    float* ob = out + ((size_t)b * 128 + h * 32) * LL + l;
    #pragma unroll
    for (int d = 0; d < 32; ++d) ob[(size_t)d * LL] = acc[d];
}

// ---------------------------------------------------------------------------
// Host orchestration.  Workspace: bufA | bufB | bufC(attn) | f16 weights ~172 MB.
// ---------------------------------------------------------------------------
extern "C" void kernel_launch(void* const* d_in, const int* in_sizes, int n_in,
                              void* d_out, int out_size, void* d_ws, size_t ws_size,
                              hipStream_t stream) {
    const float* x    = (const float*)d_in[0];
    const float* fg   = (const float*)d_in[1];
    const float* bn1g = (const float*)d_in[3],  *bn1b = (const float*)d_in[4];
    const float* bn2g = (const float*)d_in[6],  *bn2b = (const float*)d_in[7];
    const float* v_b  = (const float*)d_in[9];
    const float* at_b = (const float*)d_in[11];
    const float* pj_b = (const float*)d_in[13];
    const float* bn3g = (const float*)d_in[15], *bn3b = (const float*)d_in[16];
    const float* bn4g = (const float*)d_in[18], *bn4b = (const float*)d_in[19];

    const size_t BUF  = (size_t)BATCH * 128 * LL * sizeof(float);   // 37,748,736
    const size_t CBUF = (size_t)BATCH * 324 * LL * sizeof(float);   // 95,551,488
    char* ws = (char*)d_ws;
    float* bufA = (float*)ws;
    float* bufB = (float*)(ws + BUF);
    float* bufC = (float*)(ws + 2 * BUF);
    _Float16* w16 = (_Float16*)(ws + 2 * BUF + CBUF);

    _Float16* w1h = w16;                    // 128*576   (r-major)
    _Float16* w2h = w1h + 128 * 576;        // 128*1152  (r-major)
    _Float16* wvh = w2h + 128 * 1152;       // 128*128
    _Float16* wah = wvh + 128 * 128;        // 384*128   (rows 324..383 zero)
    _Float16* wph = wah + 384 * 128;        // 128*128
    _Float16* w3h = wph + 128 * 128;        // 128*1152  (r-major)
    _Float16* w4h = w3h + 128 * 1152;       // 128*1152  (r-major)

    auto cvtc = [&](const void* s, _Float16* d, int M, int Cin) {
        int n = M * Cin * 9;
        cvt_conv_w_kernel<<<(n + 255) / 256, 256, 0, stream>>>((const float*)s, d, M, Cin);
    };
    auto cvtp = [&](const void* s, _Float16* d, int nsrc, int ntot) {
        cvt_pad_kernel<<<(ntot + 255) / 256, 256, 0, stream>>>((const float*)s, d, nsrc, ntot);
    };
    cvtc(d_in[2],  w1h, 128, 64);
    cvtc(d_in[5],  w2h, 128, 128);
    cvtp(d_in[8],  wvh, 128 * 128, 128 * 128);
    cvtp(d_in[10], wah, 324 * 128, 384 * 128);
    cvtp(d_in[12], wph, 128 * 128, 128 * 128);
    cvtc(d_in[14], w3h, 128, 128);
    cvtc(d_in[17], w4h, 128, 128);

    const dim3 blk(256);
    const dim3 g1(LL / NB, 1, BATCH);   // M=128
    const dim3 g2(LL / NB, 3, BATCH);   // M=324 (padded weights to 384 rows)

    // conv1 + BN + ReLU : x -> bufA
    gemm_conv_kernel<9, true, true><<<g1, blk, 0, stream>>>(x, w1h, bn1g, bn1b, bufA, 64, 128);
    // conv2 + BN + ReLU : bufA -> bufB (= xf)
    gemm_conv_kernel<9, true, true><<<g1, blk, 0, stream>>>(bufA, w2h, bn2g, bn2b, bufB, 128, 128);
    // v linear : bufB -> bufA (= v)
    gemm_conv_kernel<1, false, false><<<g1, blk, 0, stream>>>(bufB, wvh, nullptr, v_b, bufA, 128, 128);
    // attention logits : fg -> bufC
    gemm_conv_kernel<1, false, false><<<g2, blk, 0, stream>>>(fg, wah, nullptr, at_b, bufC, 128, 324);
    // softmax + einsum + fold : (bufC, bufA) -> bufB (= folded)
    const int aggT = BATCH * 4 * LL;
    agg_fold_kernel<<<(aggT + 255) / 256, 256, 0, stream>>>(bufC, bufA, bufB);
    // proj linear : bufB -> bufA
    gemm_conv_kernel<1, false, false><<<g1, blk, 0, stream>>>(bufB, wph, nullptr, pj_b, bufA, 128, 128);
    // conv3 + BN + ReLU : bufA -> bufB
    gemm_conv_kernel<9, true, true><<<g1, blk, 0, stream>>>(bufA, w3h, bn3g, bn3b, bufB, 128, 128);
    // conv4 + BN + ReLU : bufB -> d_out
    gemm_conv_kernel<9, true, true><<<g1, blk, 0, stream>>>(bufB, w4h, bn4g, bn4b, (float*)d_out, 128, 128);
}
